// GPT_8864812499402
// MI455X (gfx1250) — compile-verified, matrix operand order checked by reading, not run on hardware
//
#include <hip/hip_runtime.h>

// ---------------------------------------------------------------------------
// Types for CDNA5 WMMA (wave32, v_wmma_f32_16x16x32_bf16)
// ---------------------------------------------------------------------------
typedef __bf16 bf16;
typedef bf16  bf16x8 __attribute__((ext_vector_type(8)));
typedef bf16  v16bf  __attribute__((ext_vector_type(16)));
typedef float v8f    __attribute__((ext_vector_type(8)));

// Native converts: let clang pick v_cvt_pk_bf16_f32 on gfx1250
__device__ __forceinline__ bf16 f2bf(float f) { return (bf16)f; }

// Model constants (fixed by the reference)
#define BB   4
#define SS   512
#define TT   512
#define EE   768
#define HH   12
#define HSD  64
#define VV   32000
#define LL   6
#define FFD  3072
#define MT   (BB * TT)          // 2048 rows in every activation GEMM

// GEMM tiling: block 256x64, 8 waves stacked down M, wave tile 32x64 (2x4 WMMA)
#define BM   256
#define BN   64
#define KC   32
#define KPAD 40                  // 40 bf16 = 80B row pitch: 16B aligned, bank-conflict free
#define ASZ  (BM * KPAD)
#define BSZ  (BN * KPAD)

// ---------------------------------------------------------------------------
// Fragment load from LDS, matching the CDNA5 16-bit A/B operand layout:
// lanes 0-15 hold K {0..7, 16..23}, lanes 16-31 hold K {8..15, 24..31}.
// Two ds_load_b128 per fragment.
// ---------------------------------------------------------------------------
__device__ __forceinline__ v16bf load_frag(const bf16* rowp, int kb) {
    bf16x8 lo = *(const bf16x8*)(rowp + kb);
    bf16x8 hi = *(const bf16x8*)(rowp + kb + 16);
    return __builtin_shufflevector(lo, hi, 0,1,2,3,4,5,6,7,8,9,10,11,12,13,14,15);
}

__device__ __forceinline__ v8f wmma_bf16(v16bf a, v16bf b, v8f c) {
    return __builtin_amdgcn_wmma_f32_16x16x32_bf16(false, a, false, b,
                                                   (short)0, c, false, false);
}

// ---------------------------------------------------------------------------
// Batched GEMM:  Cout[g] = act( A[g] x B[g]^(T?) + bias + Cin[g] )
//   A  : fp32 [M,K], row stride lda, batch stride sA  (bf16-converted in LDS)
//   B  : fp32; bKN=0 -> [N,K] row-major (x @ W.T); bKN=1 -> [K,N]
//   Cin: optional residual (row stride ldc), bias: optional per-column
// Double-buffered LDS, one barrier per K-chunk; global loads for chunk k+1
// overlap the 8 WMMAs of chunk k.  M%256==0, N%64==0, K%32==0 always hold.
// ---------------------------------------------------------------------------
__global__ __launch_bounds__(256)
void gemm_bf16_kernel(const float* __restrict__ A, long long sA, int lda,
                      const float* __restrict__ Bm, long long sB, int ldb, int bKN,
                      const float* __restrict__ bias,
                      const float* __restrict__ Cin, long long sCin,
                      float* __restrict__ Cout, long long sC, int ldc,
                      int M, int N, int K, int relu)
{
    __shared__ bf16 As[2 * ASZ];
    __shared__ bf16 Bs[2 * BSZ];

    const int bz = blockIdx.z;
    A    += (long long)bz * sA;
    Bm   += (long long)bz * sB;
    Cout += (long long)bz * sC;
    if (Cin) Cin += (long long)bz * sCin;

    const int tile_n = blockIdx.x * BN;
    const int tile_m = blockIdx.y * BM;
    const int tid  = threadIdx.x;
    const int lane = tid & 31;
    const int wave = tid >> 5;
    const int wm = wave << 5;             // wave m-offset: 8 waves cover 256 rows
    const int lr = lane & 15;
    const int kb = (lane < 16) ? 0 : 8;   // K sub-offset per ISA fragment layout

    v8f acc[2][4];
#pragma unroll
    for (int i = 0; i < 2; i++)
#pragma unroll
        for (int j = 0; j < 4; j++) acc[i][j] = v8f{0,0,0,0,0,0,0,0};

    // LDS-fill work assignment: A -> one full 32-col row per thread,
    //                           B -> 8 elements per thread
    const int ar = tid;
    const int br = tid >> 2;
    const int bc = (tid & 3) << 3;

    float4 a_reg[8];
    float  b_reg[8];

    auto load_global = [&](int k0) {
        const float* gp = A + (long long)(tile_m + ar) * lda + k0;
#pragma unroll
        for (int i = 0; i < 8; i++) a_reg[i] = ((const float4*)gp)[i];
        if (!bKN) {
            const float4* gb = (const float4*)(Bm + (long long)(tile_n + br) * ldb + k0 + bc);
            float4 f0 = gb[0], f1 = gb[1];
            b_reg[0] = f0.x; b_reg[1] = f0.y; b_reg[2] = f0.z; b_reg[3] = f0.w;
            b_reg[4] = f1.x; b_reg[5] = f1.y; b_reg[6] = f1.z; b_reg[7] = f1.w;
        } else {        // weight stored [K,N]: strided gather (transposed into LDS)
            const float* gp2 = Bm + (long long)(k0 + bc) * ldb + tile_n + br;
#pragma unroll
            for (int j = 0; j < 8; j++) b_reg[j] = gp2[(long long)j * ldb];
        }
    };

    auto store_lds = [&](int buf) {
        bf16* da = &As[buf * ASZ + ar * KPAD];
#pragma unroll
        for (int i = 0; i < 4; i++) {
            float4 f0 = a_reg[2 * i], f1 = a_reg[2 * i + 1];
            bf16x8 pk = { f2bf(f0.x), f2bf(f0.y), f2bf(f0.z), f2bf(f0.w),
                          f2bf(f1.x), f2bf(f1.y), f2bf(f1.z), f2bf(f1.w) };
            *(bf16x8*)(da + 8 * i) = pk;
        }
        bf16x8 pb = { f2bf(b_reg[0]), f2bf(b_reg[1]), f2bf(b_reg[2]), f2bf(b_reg[3]),
                      f2bf(b_reg[4]), f2bf(b_reg[5]), f2bf(b_reg[6]), f2bf(b_reg[7]) };
        *(bf16x8*)&Bs[buf * BSZ + br * KPAD + bc] = pb;
    };

    auto compute = [&](int buf) {
        const bf16* Ab = &As[buf * ASZ];
        const bf16* Bb = &Bs[buf * BSZ];
        v16bf a0 = load_frag(Ab + (wm      + lr) * KPAD, kb);
        v16bf a1 = load_frag(Ab + (wm + 16 + lr) * KPAD, kb);
        v16bf b0 = load_frag(Bb + (       lr) * KPAD, kb);
        v16bf b1 = load_frag(Bb + (16 + lr) * KPAD, kb);
        v16bf b2 = load_frag(Bb + (32 + lr) * KPAD, kb);
        v16bf b3 = load_frag(Bb + (48 + lr) * KPAD, kb);
        acc[0][0] = wmma_bf16(a0, b0, acc[0][0]);
        acc[0][1] = wmma_bf16(a0, b1, acc[0][1]);
        acc[0][2] = wmma_bf16(a0, b2, acc[0][2]);
        acc[0][3] = wmma_bf16(a0, b3, acc[0][3]);
        acc[1][0] = wmma_bf16(a1, b0, acc[1][0]);
        acc[1][1] = wmma_bf16(a1, b1, acc[1][1]);
        acc[1][2] = wmma_bf16(a1, b2, acc[1][2]);
        acc[1][3] = wmma_bf16(a1, b3, acc[1][3]);
    };

    // ---- software pipeline: double-buffered LDS, one barrier per chunk ----
    load_global(0);
    store_lds(0);
    __syncthreads();
    int buf = 0;
    for (int k0 = 0; k0 < K; k0 += KC) {
        const bool has_next = (k0 + KC) < K;
        if (has_next) {
            load_global(k0 + KC);                       // in flight during WMMAs
            if (k0 + 2 * KC < K)
                __builtin_prefetch(A + (long long)(tile_m + ar) * lda + k0 + 2 * KC, 0, 1);
        }
        compute(buf);
        if (has_next) store_lds(buf ^ 1);
        __syncthreads();
        buf ^= 1;
    }

    // ---- epilogue: bias + residual + relu; D layout: lane=col, VGPR r=row(+8) ----
    const int m0 = tile_m + wm + ((lane < 16) ? 0 : 8);
    const int n0 = tile_n + lr;
#pragma unroll
    for (int i = 0; i < 2; i++) {
#pragma unroll
        for (int j = 0; j < 4; j++) {
            const int mb = m0 + (i << 4);
            const int n  = n0 + (j << 4);
            const float bv = bias ? bias[n] : 0.0f;
#pragma unroll
            for (int r = 0; r < 8; r++) {
                long long off = (long long)(mb + r) * ldc + n;
                float v = acc[i][j][r] + bv;
                if (Cin)  v += Cin[off];
                if (relu) v = fmaxf(v, 0.0f);
                Cout[off] = v;
            }
        }
    }
}

// ---------------------------------------------------------------------------
// Block-wide reduction (256 threads = 8 wave32)
// ---------------------------------------------------------------------------
template <bool MAXOP>
__device__ __forceinline__ float block_reduce(float v, float* red, int tid) {
#pragma unroll
    for (int o = 16; o > 0; o >>= 1) {
        float w = __shfl_xor(v, o, 32);
        v = MAXOP ? fmaxf(v, w) : (v + w);
    }
    if ((tid & 31) == 0) red[tid >> 5] = v;
    __syncthreads();
    float r = red[0];
#pragma unroll
    for (int i = 1; i < 8; i++) r = MAXOP ? fmaxf(r, red[i]) : (r + red[i]);
    __syncthreads();
    return r;
}

__global__ __launch_bounds__(256)
void layernorm_kernel(const float* __restrict__ x, const float* __restrict__ w,
                      const float* __restrict__ b, float* __restrict__ out, int E)
{
    __shared__ float red[8];
    long long row = blockIdx.x;
    const float* xr = x + row * E;
    float* orow = out + row * E;
    int tid = threadIdx.x;
    float s = 0.f, ss = 0.f;
    for (int e = tid; e < E; e += 256) { float v = xr[e]; s += v; ss += v * v; }
    s  = block_reduce<false>(s,  red, tid);
    ss = block_reduce<false>(ss, red, tid);
    float mu  = s / (float)E;
    float var = ss / (float)E - mu * mu;
    float rstd = rsqrtf(var + 1e-5f);
    for (int e = tid; e < E; e += 256)
        orow[e] = (xr[e] - mu) * rstd * w[e] + b[e];
}

__global__ __launch_bounds__(256)
void softmax_kernel(float* __restrict__ sc, int S, int T, int causal, float scale)
{
    __shared__ float red[8];
    long long row = blockIdx.x;            // over H*B*T
    float* p = sc + row * (long long)S;
    const int t  = (int)(row % T);
    const int tid = threadIdx.x;
    float vals[2];
    float mx = -3.0e38f;
#pragma unroll
    for (int i = 0; i < 2; i++) {
        int s = tid + i * 256;
        float v = p[s] * scale;
        if (causal && s > t) v = -3.0e38f;
        vals[i] = v;
        mx = fmaxf(mx, v);
    }
    mx = block_reduce<true>(mx, red, tid);
    float sum = 0.f;
#pragma unroll
    for (int i = 0; i < 2; i++) {
        int s = tid + i * 256;
        float e = __expf(vals[i] - mx);
        if (causal && s > t) e = 0.f;
        vals[i] = e;
        sum += e;
    }
    sum = block_reduce<false>(sum, red, tid);
    float inv = 1.0f / sum;
#pragma unroll
    for (int i = 0; i < 2; i++) p[tid + i * 256] = vals[i] * inv;
}

__global__ __launch_bounds__(256)
void embed_kernel(const int* __restrict__ ids, const float* __restrict__ tok,
                  const float* __restrict__ pos, float* __restrict__ x,
                  int Slen, int E)
{
    long long m = blockIdx.x;
    int id = ids[m];
    int pp = (int)(m % Slen);
    const float* tr = tok + (long long)id * E;
    const float* pr = pos + (long long)pp * E;
    float* xr = x + m * E;
    for (int e = threadIdx.x; e < E; e += 256) xr[e] = tr[e] + pr[e];
}

// ---------------------------------------------------------------------------
// Host orchestration
// ---------------------------------------------------------------------------
static inline void launch_gemm(hipStream_t st,
    const float* A, long long sA, int lda,
    const float* B, long long sB, int ldb, int bKN,
    const float* bias, const float* Cin, long long sCin,
    float* Cout, long long sC, int ldc,
    int M, int N, int K, int relu, int batch)
{
    dim3 grid((unsigned)(N / BN), (unsigned)(M / BM), (unsigned)batch);
    gemm_bf16_kernel<<<grid, dim3(256), 0, st>>>(A, sA, lda, B, sB, ldb, bKN,
                                                 bias, Cin, sCin, Cout, sC, ldc,
                                                 M, N, K, relu);
}

extern "C" void kernel_launch(void* const* d_in, const int* in_sizes, int n_in,
                              void* d_out, int out_size, void* d_ws, size_t ws_size,
                              hipStream_t stream)
{
    if (n_in < 39) return;   // expects dict-insertion-order flattening

    const int*   enc_input = (const int*)d_in[0];
    const int*   dec_input = (const int*)d_in[1];
    const float* tok_emb   = (const float*)d_in[2];
    const float* pos_emb   = (const float*)d_in[3];
    // enc params (13): ln1_w ln1_b ln2_w ln2_b sa_q sa_k sa_v sa_pw sa_pb w1 b1 w2 b2
    const float* const* ep = (const float* const*)&d_in[4];
    // dec params (20): enc order + ln3_w ln3_b ca_q ca_k ca_v ca_pw ca_pb
    const float* const* dp = (const float* const*)&d_in[17];
    const float* lm_w = (const float*)d_in[37];
    const float* lm_b = (const float*)d_in[38];
    float* logits = (float*)d_out;

    // ---- workspace carve (all fp32) ----
    float* p = (float*)d_ws;
    auto alloc = [&](size_t n) { float* r = p; p += n; return r; };
    float* x_enc = alloc((size_t)BB * SS * EE);
    float* x_dec = alloc((size_t)BB * TT * EE);
    float* hbuf  = alloc((size_t)BB * TT * EE);
    float* qb    = alloc((size_t)BB * TT * EE);
    float* kbuf  = alloc((size_t)BB * SS * EE);
    float* vbuf  = alloc((size_t)BB * SS * EE);
    float* ctx   = alloc((size_t)BB * TT * EE);
    float* sc    = alloc((size_t)HH * BB * TT * SS);
    float* ffh   = alloc((size_t)BB * TT * FFD);
    if ((size_t)((char*)p - (char*)d_ws) > ws_size) return;

    const long long WQKV = (long long)EE * HSD;         // per-head weight stride
    const float scale = 0.125f;                          // HS^-0.5

    // attention block: q from q_src, k/v from kv_src; writes x += proj(attn)+pb
    auto run_attn = [&](const float* q_src, const float* kv_src,
                        const float* wq, const float* wk, const float* wv,
                        const float* pw, const float* pb, float* x, int causal)
    {
        // QKV per head: [MT,E] x [E,HS] -> columns h*HS..h*HS+63 ([b,t,h,k] layout)
        launch_gemm(stream, q_src, 0, EE, wq, WQKV, HSD, 1, nullptr, nullptr, 0,
                    qb,   HSD, EE, MT, HSD, EE, 0, HH);
        launch_gemm(stream, kv_src, 0, EE, wk, WQKV, HSD, 1, nullptr, nullptr, 0,
                    kbuf, HSD, EE, MT, HSD, EE, 0, HH);
        launch_gemm(stream, kv_src, 0, EE, wv, WQKV, HSD, 1, nullptr, nullptr, 0,
                    vbuf, HSD, EE, MT, HSD, EE, 0, HH);
        // scores[g] = Q[g] K[g]^T  (g = h*B+b via flat [H,B,T,HS] reinterpretation)
        launch_gemm(stream, qb, (long long)TT * HSD, HSD,
                    kbuf, (long long)SS * HSD, HSD, 0, nullptr, nullptr, 0,
                    sc, (long long)TT * SS, SS, TT, SS, HSD, 0, HH * BB);
        softmax_kernel<<<HH * BB * TT, 256, 0, stream>>>(sc, SS, TT, causal, scale);
        // ctx[g] = P[g] V[g]  -> flat [H,B,T,HS], read back as [B*T, E]
        launch_gemm(stream, sc, (long long)TT * SS, SS,
                    vbuf, (long long)SS * HSD, HSD, 1, nullptr, nullptr, 0,
                    ctx, (long long)TT * HSD, HSD, TT, HSD, SS, 0, HH * BB);
        // x = x + ctx @ pw.T + pb
        launch_gemm(stream, ctx, 0, EE, pw, 0, EE, 0, pb, x, 0,
                    x, 0, EE, MT, EE, EE, 0, 1);
    };

    // ======================= encoder =======================
    embed_kernel<<<BB * SS, 256, 0, stream>>>(enc_input, tok_emb, pos_emb, x_enc, SS, EE);
    for (int l = 0; l < LL; l++) {
        layernorm_kernel<<<MT, 256, 0, stream>>>(x_enc, ep[0] + l * EE, ep[1] + l * EE, hbuf, EE);
        run_attn(hbuf, hbuf,
                 ep[4] + (long long)l * HH * WQKV, ep[5] + (long long)l * HH * WQKV,
                 ep[6] + (long long)l * HH * WQKV,
                 ep[7] + (long long)l * EE * EE, ep[8] + l * EE, x_enc, 0);
        layernorm_kernel<<<MT, 256, 0, stream>>>(x_enc, ep[2] + l * EE, ep[3] + l * EE, hbuf, EE);
        launch_gemm(stream, hbuf, 0, EE, ep[9] + (long long)l * FFD * EE, 0, EE, 0,
                    ep[10] + l * FFD, nullptr, 0, ffh, 0, FFD, MT, FFD, EE, 1, 1);
        launch_gemm(stream, ffh, 0, FFD, ep[11] + (long long)l * EE * FFD, 0, FFD, 0,
                    ep[12] + l * EE, x_enc, 0, x_enc, 0, EE, MT, EE, FFD, 0, 1);
    }

    // ======================= decoder =======================
    embed_kernel<<<BB * TT, 256, 0, stream>>>(dec_input, tok_emb, pos_emb, x_dec, TT, EE);
    for (int l = 0; l < LL; l++) {
        // masked self-attention
        layernorm_kernel<<<MT, 256, 0, stream>>>(x_dec, dp[0] + l * EE, dp[1] + l * EE, hbuf, EE);
        run_attn(hbuf, hbuf,
                 dp[4] + (long long)l * HH * WQKV, dp[5] + (long long)l * HH * WQKV,
                 dp[6] + (long long)l * HH * WQKV,
                 dp[7] + (long long)l * EE * EE, dp[8] + l * EE, x_dec, 1);
        // cross-attention (q from ln2(x), k/v from raw encoder output)
        layernorm_kernel<<<MT, 256, 0, stream>>>(x_dec, dp[2] + l * EE, dp[3] + l * EE, hbuf, EE);
        run_attn(hbuf, x_enc,
                 dp[15] + (long long)l * HH * WQKV, dp[16] + (long long)l * HH * WQKV,
                 dp[17] + (long long)l * HH * WQKV,
                 dp[18] + (long long)l * EE * EE, dp[19] + l * EE, x_dec, 0);
        // ffn
        layernorm_kernel<<<MT, 256, 0, stream>>>(x_dec, dp[13] + l * EE, dp[14] + l * EE, hbuf, EE);
        launch_gemm(stream, hbuf, 0, EE, dp[9] + (long long)l * FFD * EE, 0, EE, 0,
                    dp[10] + l * FFD, nullptr, 0, ffh, 0, FFD, MT, FFD, EE, 1, 1);
        launch_gemm(stream, ffh, 0, FFD, dp[11] + (long long)l * EE * FFD, 0, FFD, 0,
                    dp[12] + l * EE, x_dec, 0, x_dec, 0, EE, MT, EE, FFD, 0, 1);
    }

    // ======================= LM head =======================
    launch_gemm(stream, x_dec, 0, EE, lm_w, 0, EE, 0, lm_b, nullptr, 0,
                logits, 0, VV, MT, VV, EE, 0, 1);
}
